// CausalSelfAttention_32762010534335
// MI455X (gfx1250) — matrix operand order repaired and checked by reading, not compiled, not run on hardware
//
#include <hip/hip_runtime.h>
#include <hip/hip_bf16.h>
#include <cstdint>

// ---------------------------------------------------------------------------
// CausalSelfAttention forward for MI455X (gfx1250, wave32)
//   - all matmuls via v_wmma_f32_16x16x32_f16 (fp32 accum)
//   - K/V tiles staged with global_load_async_to_lds_b128 (ASYNCcnt),
//     double-buffered so copy(t+1) overlaps WMMA compute(t)
// B=2, T=2048, C=2048, H=16, KV=4, D=128
// ---------------------------------------------------------------------------

#define B_   2
#define T_   2048
#define C_   2048
#define H_   16
#define KV_  4
#define D_   128
#define KVC_ 512
#define M_   (B_ * T_)                  // 4096 rows
#define SCALE_ 0.08838834764831845f     // 1/sqrt(128)

typedef _Float16 h8   __attribute__((ext_vector_type(8)));
typedef _Float16 v16h __attribute__((ext_vector_type(16)));
typedef float    v8f  __attribute__((ext_vector_type(8)));

static __device__ __forceinline__ v16h cat16(h8 lo, h8 hi) {
    return __builtin_shufflevector(lo, hi, 0,1,2,3,4,5,6,7,8,9,10,11,12,13,14,15);
}

static __device__ __forceinline__ v8f wmma_f16(v16h a, v16h b, v8f c) {
    // D = A(16x32 f16) * B(32x16 f16) + C(16x16 f32)
    return __builtin_amdgcn_wmma_f32_16x16x32_f16(
        /*neg_a=*/false, a, /*neg_b=*/false, b,
        /*c_mod=*/(short)0, c, /*reuse_a=*/false, /*reuse_b=*/false);
}

// 16B-per-lane async global->LDS copy (ASYNCcnt). VDST = LDS byte offset
// (low 32 bits of the generic pointer), VADDR = 64-bit global address.
static __device__ __forceinline__ void async_cp_b128(const void* g, void* l) {
    asm volatile("global_load_async_to_lds_b128 %0, %1, off"
                 :: "v"((uint32_t)(size_t)l), "v"(g) : "memory");
}
static __device__ __forceinline__ void async_cp_b64(const void* g, void* l) {
    asm volatile("global_load_async_to_lds_b64 %0, %1, off"
                 :: "v"((uint32_t)(size_t)l), "v"(g) : "memory");
}
static __device__ __forceinline__ void wait_async0() {
    asm volatile("s_wait_asynccnt 0" ::: "memory");
}
static __device__ __forceinline__ void wait_ds0() {
    asm volatile("s_wait_dscnt 0" ::: "memory");
}

#define ZERO8 ((v8f){0.f,0.f,0.f,0.f,0.f,0.f,0.f,0.f})

// ---------------------------------------------------------------------------
// Generic tiled GEMM:  C[M,N] = A[M,K] * B[K,N]   (row-major, fp32 out)
// A fp32 (AHALF=false, converted during staging) or f16 (AHALF=true, staged
// with async global->LDS copies). Block tile 64x64, BK=32, 128 threads.
// ---------------------------------------------------------------------------
template<bool AHALF>
__global__ void __launch_bounds__(128)
gemm_wmma_f16(const void* __restrict__ Av, const float* __restrict__ Bm,
              float* __restrict__ Cm, int M, int N, int K) {
    __shared__ _Float16 As[64][40];     // 32 data cols + pad (80B rows)
    __shared__ _Float16 Bs[32][72];     // 64 data cols + pad (144B rows)

    const int n0   = blockIdx.x * 64;
    const int m0   = blockIdx.y * 64;
    const int tid  = threadIdx.x;
    const int wave = tid >> 5;
    const int lane = tid & 31;
    const int n    = lane & 15;         // C/D: N index
    const int hh   = lane >> 4;         // C/D: M half

    v8f acc[4] = {ZERO8, ZERO8, ZERO8, ZERO8};

    const int arow0 = tid >> 3;         // 0..15
    const int acol  = (tid & 7) * 4;    // 0..28
    const int brow0 = tid >> 4;         // 0..7
    const int bcol  = (tid & 15) * 4;   // 0..60

    for (int k0 = 0; k0 < K; k0 += 32) {
        __syncthreads();
        // ---- stage A tile (64x32) ----
        if (AHALF) {
            const _Float16* A = (const _Float16*)Av;
            #pragma unroll
            for (int it = 0; it < 4; ++it) {
                const int row = arow0 + it * 16;
                async_cp_b64(A + (size_t)(m0 + row) * K + k0 + acol, &As[row][acol]);
            }
        } else {
            const float* A = (const float*)Av;
            #pragma unroll
            for (int it = 0; it < 4; ++it) {
                const int row = arow0 + it * 16;
                float4 a = *(const float4*)(A + (size_t)(m0 + row) * K + k0 + acol);
                As[row][acol + 0] = (_Float16)a.x;
                As[row][acol + 1] = (_Float16)a.y;
                As[row][acol + 2] = (_Float16)a.z;
                As[row][acol + 3] = (_Float16)a.w;
            }
        }
        // ---- stage B tile (32x64), fp32 -> f16 ----
        #pragma unroll
        for (int it = 0; it < 4; ++it) {
            const int row = brow0 + it * 8;
            float4 b = *(const float4*)(Bm + (size_t)(k0 + row) * N + n0 + bcol);
            Bs[row][bcol + 0] = (_Float16)b.x;
            Bs[row][bcol + 1] = (_Float16)b.y;
            Bs[row][bcol + 2] = (_Float16)b.z;
            Bs[row][bcol + 3] = (_Float16)b.w;
        }
        if (AHALF) wait_async0();
        __syncthreads();

        // A fragment: row = wave*16 + lane%16, K = hh*8+{0..7}, hh*8+16+{0..7}
        const int ar = wave * 16 + n;
        h8 alo = *(const h8*)&As[ar][hh * 8];
        h8 ahi = *(const h8*)&As[ar][hh * 8 + 16];
        v16h af = cat16(alo, ahi);

        #pragma unroll
        for (int j = 0; j < 4; ++j) {
            v16h bf;
            #pragma unroll
            for (int s = 0; s < 16; ++s) bf[s] = Bs[hh * 8 + (s & 7) + (s >> 3) * 16][j * 16 + n];
            acc[j] = wmma_f16(af, bf, acc[j]);
        }
    }

    #pragma unroll
    for (int j = 0; j < 4; ++j)
        #pragma unroll
        for (int i = 0; i < 8; ++i) {
            const int row = m0 + wave * 16 + i + 8 * hh;
            Cm[(size_t)row * N + n0 + j * 16 + n] = acc[j][i];
        }
}

// ---------------------------------------------------------------------------
// RoPE + RMS-norm on q,k; relayout to f16:
//   q  -> (B, H, T, D)   (row-major per head, for Q fragments)
//   k  -> (B, KV, D, T)  (d-major! so attention K staging is contiguous)
//   v  -> (B, KV, T, D)
// ---------------------------------------------------------------------------
__global__ void __launch_bounds__(256)
rope_rms_relayout(const float* __restrict__ qf, const float* __restrict__ kf,
                  const float* __restrict__ vf,
                  const float* __restrict__ cosp, const float* __restrict__ sinp,
                  _Float16* __restrict__ qh, _Float16* __restrict__ kd,
                  _Float16* __restrict__ vh) {
    const int bt   = blockIdx.x;
    const int b    = bt / T_;
    const int t    = bt % T_;
    const int tid  = threadIdx.x;
    const int wave = tid >> 5;
    const int lane = tid & 31;

    const float ca = cosp[t * 64 + lane];
    const float cb = cosp[t * 64 + lane + 32];
    const float sa = sinp[t * 64 + lane];
    const float sb = sinp[t * 64 + lane + 32];

    for (int u = wave; u < H_ + KV_; u += 8) {
        const bool isq = (u < H_);
        const int  h   = isq ? u : (u - H_);
        const float* src = isq ? (qf + (size_t)bt * C_   + h * D_)
                               : (kf + (size_t)bt * KVC_ + h * D_);
        const float x1a = src[lane];
        const float x1b = src[lane + 32];
        const float x2a = src[lane + 64];
        const float x2b = src[lane + 96];
        const float o1a =  x1a * ca + x2a * sa;
        const float o1b =  x1b * cb + x2b * sb;
        const float o2a = -x1a * sa + x2a * ca;
        const float o2b = -x1b * sb + x2b * cb;

        float ss = o1a * o1a + o1b * o1b + o2a * o2a + o2b * o2b;
        #pragma unroll
        for (int m = 1; m < 32; m <<= 1) ss += __shfl_xor(ss, m, 32);
        const float r = rsqrtf(ss * (1.0f / 128.0f) + 1.1920929e-7f);

        if (isq) {
            _Float16* dst = qh + (((size_t)(b * H_ + h) * T_) + t) * D_;
            dst[lane]      = (_Float16)(o1a * r);
            dst[lane + 32] = (_Float16)(o1b * r);
            dst[lane + 64] = (_Float16)(o2a * r);
            dst[lane + 96] = (_Float16)(o2b * r);
        } else {
            // d-major: kd[((b*KV + h)*D + d)*T + t]
            _Float16* dst = kd + ((size_t)(b * KV_ + h) * D_) * T_ + t;
            dst[(size_t)(lane)       * T_] = (_Float16)(o1a * r);
            dst[(size_t)(lane + 32)  * T_] = (_Float16)(o1b * r);
            dst[(size_t)(lane + 64)  * T_] = (_Float16)(o2a * r);
            dst[(size_t)(lane + 96)  * T_] = (_Float16)(o2b * r);
        }
    }

    // v: convert + relayout (B, KV, T, D)
    for (int c = tid; c < KVC_; c += 256) {
        const int kvh = c >> 7, d = c & 127;
        vh[(((size_t)(b * KV_ + kvh) * T_) + t) * D_ + d] =
            (_Float16)vf[(size_t)bt * KVC_ + c];
    }
}

// ---------------------------------------------------------------------------
// Flash attention (causal, GQA rep=4). 128 threads = 4 waves; each wave owns
// a 16-row q tile. KV streamed in 32-row tiles, double-buffered in LDS via
// async global->LDS b128 copies (copy t+1 overlaps WMMA compute of t).
// ---------------------------------------------------------------------------
__global__ void __launch_bounds__(128)
flash_attn_wmma(const _Float16* __restrict__ qh, const _Float16* __restrict__ kd,
                const _Float16* __restrict__ vh, _Float16* __restrict__ yh) {
    __shared__ _Float16 Kt[2][D_][40];    // K d-major tile: [d][kv_local] + pad
    __shared__ _Float16 Vs[2][32][136];   // V row-major:    [kv_local][d] + pad
    __shared__ _Float16 Ps[4][16][40];    // per-wave P repack + pad

    const int qtile = blockIdx.x;
    const int h     = blockIdx.y;
    const int b     = blockIdx.z;
    const int kvh   = h >> 2;             // GQA: 4 q-heads per kv-head
    const int qbase = qtile * 64;

    const int tid  = threadIdx.x;
    const int wave = tid >> 5;
    const int lane = tid & 31;
    const int n    = lane & 15;
    const int hh   = lane >> 4;

    // ---- preload Q fragments (16 rows x 128 d) straight from global ----
    const _Float16* qptr =
        qh + (((size_t)(b * H_ + h) * T_) + qbase + wave * 16 + n) * D_;
    v16h qa[4];
    #pragma unroll
    for (int c = 0; c < 4; ++c) {
        h8 lo = *(const h8*)(qptr + c * 32 + hh * 8);
        h8 hi = *(const h8*)(qptr + c * 32 + hh * 8 + 16);
        qa[c] = cat16(lo, hi);
    }

    v8f acc[8] = {ZERO8, ZERO8, ZERO8, ZERO8, ZERO8, ZERO8, ZERO8, ZERO8};
    float mrow[8], lrow[8];
    #pragma unroll
    for (int i = 0; i < 8; ++i) { mrow[i] = -3.0e38f; lrow[i] = 0.0f; }

    const _Float16* kdbase = kd + ((size_t)(b * KV_ + kvh) * D_) * T_;
    const _Float16* vbase  = vh + ((size_t)(b * KV_ + kvh) * T_) * D_;

    const int srow   = tid >> 4;          // 0..7
    const int sdcol  = (tid & 15) * 8;    // 0..120
    const int ntiles = (qbase + 64) >> 5; // causal bound / 32

    // async stage of one 32-wide KV tile into buffer bi
    auto stage = [&](int kv0, int bi) {
        // K: thread owns d = tid; 64B contiguous along t in d-major layout
        const _Float16* ks = kdbase + (size_t)tid * T_ + kv0;
        #pragma unroll
        for (int j = 0; j < 4; ++j)
            async_cp_b128(ks + 8 * j, &Kt[bi][tid][8 * j]);
        // V: row-major 32x128 tile
        #pragma unroll
        for (int it = 0; it < 4; ++it) {
            const int r = srow + it * 8;
            async_cp_b128(vbase + (size_t)(kv0 + r) * D_ + sdcol, &Vs[bi][r][sdcol]);
        }
    };

    stage(0, 0);
    wait_async0();
    __syncthreads();

    for (int t = 0; t < ntiles; ++t) {
        const int kv0 = t * 32;
        const int bi  = t & 1;
        if (t + 1 < ntiles) stage(kv0 + 32, bi ^ 1);   // overlap copy with compute

        // ---- S = Q * K^T  (two 16x16 tiles over 32 kv cols) ----
        v8f S0 = ZERO8, S1 = ZERO8;
        #pragma unroll
        for (int c = 0; c < 4; ++c) {
            v16h b0, b1;
            #pragma unroll
            for (int s = 0; s < 16; ++s) {
                const int d = c * 32 + hh * 8 + (s & 7) + (s >> 3) * 16;
                b0[s] = Kt[bi][d][n];
                b1[s] = Kt[bi][d][n + 16];
            }
            S0 = wmma_f16(qa[c], b0, S0);
            S1 = wmma_f16(qa[c], b1, S1);
        }

        // ---- online softmax (row = qbase+wave*16+i+8*hh, col = kv0+n[+16]) ----
        #pragma unroll
        for (int i = 0; i < 8; ++i) {
            const int row = qbase + wave * 16 + i + 8 * hh;
            const int c0  = kv0 + n;
            float s0 = (c0      <= row) ? S0[i] * SCALE_ : -3.0e38f;
            float s1 = (c0 + 16 <= row) ? S1[i] * SCALE_ : -3.0e38f;
            float mx = fmaxf(s0, s1);
            #pragma unroll
            for (int m = 1; m < 16; m <<= 1) mx = fmaxf(mx, __shfl_xor(mx, m, 32));
            const float mn   = fmaxf(mrow[i], mx);
            const float corr = __expf(mrow[i] - mn);
            const float p0   = __expf(s0 - mn);
            const float p1   = __expf(s1 - mn);
            float sum = p0 + p1;
            #pragma unroll
            for (int m = 1; m < 16; m <<= 1) sum += __shfl_xor(sum, m, 32);
            lrow[i] = lrow[i] * corr + sum;
            mrow[i] = mn;
            #pragma unroll
            for (int j = 0; j < 8; ++j) acc[j][i] *= corr;
            const int ri = i + 8 * hh;
            Ps[wave][ri][n]      = (_Float16)p0;
            Ps[wave][ri][n + 16] = (_Float16)p1;
        }
        wait_ds0();   // wave-private repack: all lanes' DS stores before reload

        // ---- P fragment (A-layout) and  acc += P * V ----
        h8 plo = *(const h8*)&Ps[wave][n][hh * 8];
        h8 phi = *(const h8*)&Ps[wave][n][hh * 8 + 16];
        v16h pf = cat16(plo, phi);
        #pragma unroll
        for (int j = 0; j < 8; ++j) {
            v16h vf;
            #pragma unroll
            for (int s = 0; s < 16; ++s) vf[s] = Vs[bi][hh * 8 + (s & 7) + (s >> 3) * 16][j * 16 + n];
            acc[j] = wmma_f16(pf, vf, acc[j]);
        }

        wait_async0();     // next tile's copies have landed
        __syncthreads();   // all waves done with buffers before reuse
    }

    // ---- normalize and write y (B, T, H*D) as f16 for the output GEMM ----
    #pragma unroll
    for (int i = 0; i < 8; ++i) {
        const float inv = 1.0f / lrow[i];
        const int row = qbase + wave * 16 + i + 8 * hh;
        #pragma unroll
        for (int j = 0; j < 8; ++j) {
            yh[((size_t)(b * T_ + row)) * C_ + h * D_ + j * 16 + n] =
                (_Float16)(acc[j][i] * inv);
        }
    }
}

// ---------------------------------------------------------------------------
// Launch: qkv GEMMs -> rope/rms/relayout -> flash attention -> out GEMM
// ---------------------------------------------------------------------------
extern "C" void kernel_launch(void* const* d_in, const int* in_sizes, int n_in,
                              void* d_out, int out_size, void* d_ws, size_t ws_size,
                              hipStream_t stream) {
    const float* x    = (const float*)d_in[0];
    const float* cosp = (const float*)d_in[1];
    const float* sinp = (const float*)d_in[2];
    const float* Wq   = (const float*)d_in[3];
    const float* Wk   = (const float*)d_in[4];
    const float* Wv   = (const float*)d_in[5];
    const float* Wo   = (const float*)d_in[6];

    char* ws = (char*)d_ws;
    size_t off = 0;
    float*     qf = (float*)(ws + off);     off += (size_t)M_ * C_   * 4;  // 32 MB
    float*     kf = (float*)(ws + off);     off += (size_t)M_ * KVC_ * 4;  //  8 MB
    float*     vf = (float*)(ws + off);     off += (size_t)M_ * KVC_ * 4;  //  8 MB
    _Float16*  qh = (_Float16*)(ws + off);  off += (size_t)M_ * C_   * 2;  // 16 MB
    _Float16*  kd = (_Float16*)(ws + off);  off += (size_t)M_ * KVC_ * 2;  //  4 MB
    _Float16*  vh = (_Float16*)(ws + off);  off += (size_t)M_ * KVC_ * 2;  //  4 MB
    _Float16*  yh = (_Float16*)(ws + off);  off += (size_t)M_ * C_   * 2;  // 16 MB

    const dim3 blk(128);
    // QKV projections
    gemm_wmma_f16<false><<<dim3(C_   / 64, M_ / 64), blk, 0, stream>>>(x, Wq, qf, M_, C_,   C_);
    gemm_wmma_f16<false><<<dim3(KVC_ / 64, M_ / 64), blk, 0, stream>>>(x, Wk, kf, M_, KVC_, C_);
    gemm_wmma_f16<false><<<dim3(KVC_ / 64, M_ / 64), blk, 0, stream>>>(x, Wv, vf, M_, KVC_, C_);
    // RoPE + RMS-norm + relayout (k goes d-major for contiguous async staging)
    rope_rms_relayout<<<dim3(B_ * T_), dim3(256), 0, stream>>>(qf, kf, vf, cosp, sinp, qh, kd, vh);
    // Flash attention (async double-buffered KV tiles)
    flash_attn_wmma<<<dim3(T_ / 64, H_, B_), blk, 0, stream>>>(qh, kd, vh, yh);
    // Output projection (f16 A staged via async copies)
    gemm_wmma_f16<true><<<dim3(C_ / 64, M_ / 64), blk, 0, stream>>>(yh, Wo, (float*)d_out, M_, C_, C_);
}
